// OptimizedBEMModule_57226144252251
// MI455X (gfx1250) — matrix operand order repaired and checked by reading, not compiled, not run on hardware
//
#include <hip/hip_runtime.h>
#include <hip/hip_bf16.h>

// ---------------------------------------------------------------------------
// Problem constants (fixed by the reference): B=4, S=4096, D_IN=D_OUT=4096, R=8
// M = B*S = 16384 rows, N = D_OUT = 4096 cols, K = D_IN = 4096
// out[m,n] = sum_k x[m,k]*W[n,k] + b[n] + 2 * sum_r (x@V)[m,r]*codes[m,r]*U[n,r]
// ---------------------------------------------------------------------------

#define M_DIM 16384
#define N_DIM 4096
#define K_DIM 4096
#define R_DIM 8

#define BM 128
#define BN 128

// fallback (inline-convert) kernel tiling
#define BK 32
#define LDT 40          // bf16 stride: 80 B, multiple of 16 B

// fast (pre-converted bf16) kernel tiling
#define BK2 64
#define LDT2 72         // bf16 stride: 144 B, multiple of 16 B, conflict-free

typedef __attribute__((ext_vector_type(16))) __bf16  v16bf;
typedef __attribute__((ext_vector_type(8)))  float   v8f;
typedef __attribute__((ext_vector_type(4)))  unsigned int uint4v;

union FragU {
    v16bf  bf;
    uint4v q[2];
};

// round-to-nearest-even fp32 -> bf16, packed pair into one dword
__device__ __forceinline__ unsigned int pack_bf16(float a, float b) {
    unsigned int ua = __float_as_uint(a);
    unsigned int ub = __float_as_uint(b);
    unsigned int ha = (ua + 0x7FFFu + ((ua >> 16) & 1u)) >> 16;
    unsigned int hb = (ub + 0x7FFFu + ((ub >> 16) & 1u)) >> 16;
    return (ha & 0xFFFFu) | (hb << 16);
}

// ---------------------------------------------------------------------------
// CDNA5 async global->LDS staging (no VGPR data path, tracked by ASYNCcnt).
// INST_OFFSET is added to BOTH the global and LDS address (ISA 08 §4.4),
// so one lds-addr VGPR + one 64-bit global-addr VGPR pair covers 64 B.
// ---------------------------------------------------------------------------
__device__ __forceinline__ void async_stage64B(const void* g, void* lds) {
    unsigned long long ga = (unsigned long long)(uintptr_t)g;
    unsigned la = (unsigned)(uintptr_t)lds;   // low 32 bits = LDS byte address
    asm volatile(
        "global_load_async_to_lds_b128 %0, %1, off\n\t"
        "global_load_async_to_lds_b128 %0, %1, off offset:16\n\t"
        "global_load_async_to_lds_b128 %0, %1, off offset:32\n\t"
        "global_load_async_to_lds_b128 %0, %1, off offset:48"
        :: "v"(la), "v"(ga) : "memory");
}

__device__ __forceinline__ void wait_async0() {
    asm volatile("s_wait_asynccnt 0x0" ::: "memory");
}

// ---------------------------------------------------------------------------
// Kernel 0: bulk fp32 -> bf16 conversion (one pass; 8 elements / thread)
// ---------------------------------------------------------------------------
__global__ void __launch_bounds__(256)
cvt_bf16_kernel(const float* __restrict__ src, unsigned short* __restrict__ dst,
                long n8 /* n/8 */)
{
    long i = (long)blockIdx.x * 256 + threadIdx.x;
    if (i >= n8) return;
    const float4* sp = (const float4*)(src + i * 8);
    float4 f0 = sp[0], f1 = sp[1];
    uint4v q;
    q.x = pack_bf16(f0.x, f0.y);
    q.y = pack_bf16(f0.z, f0.w);
    q.z = pack_bf16(f1.x, f1.y);
    q.w = pack_bf16(f1.z, f1.w);
    *(uint4v*)(dst + i * 8) = q;
}

// ---------------------------------------------------------------------------
// Kernel 1: h[m,r] = 2 * codes[m,r] * sum_k x[m,k] * V[k,r]      (M x 8)
// ---------------------------------------------------------------------------
__global__ void __launch_bounds__(256)
hgen_kernel(const float* __restrict__ x, const float* __restrict__ codes,
            const float* __restrict__ V, float* __restrict__ h)
{
    const int m = blockIdx.x;
    const int t = threadIdx.x;
    const float* xr = x + (size_t)m * K_DIM;

    float s[R_DIM];
#pragma unroll
    for (int r = 0; r < R_DIM; ++r) s[r] = 0.0f;

    for (int k = t; k < K_DIM; k += 256) {
        const float xv = xr[k];
        const float4* vp = (const float4*)(V + (size_t)k * R_DIM);
        float4 v0 = vp[0];
        float4 v1 = vp[1];
        s[0] += xv * v0.x;  s[1] += xv * v0.y;
        s[2] += xv * v0.z;  s[3] += xv * v0.w;
        s[4] += xv * v1.x;  s[5] += xv * v1.y;
        s[6] += xv * v1.z;  s[7] += xv * v1.w;
    }

    __shared__ float red[R_DIM][256];
#pragma unroll
    for (int r = 0; r < R_DIM; ++r) red[r][t] = s[r];
    __syncthreads();

    for (int off = 128; off > 0; off >>= 1) {
        if (t < off) {
#pragma unroll
            for (int r = 0; r < R_DIM; ++r) red[r][t] += red[r][t + off];
        }
        __syncthreads();
    }

    if (t < R_DIM) {
        h[(size_t)m * R_DIM + t] = red[t][0] * codes[(size_t)m * R_DIM + t] * 2.0f;
    }
}

// ---------------------------------------------------------------------------
// shared epilogue: + bias[n] + dot(h[m,:], U[n,:]), store fp32
// ---------------------------------------------------------------------------
__device__ __forceinline__ void gemm_epilogue(
    const v8f acc[2][4], const float* __restrict__ bias,
    const float* __restrict__ U, const float* __restrict__ h,
    float* __restrict__ out,
    int m0, int n0, int wm, int wn, int l16, int hihalf)
{
    float4 u0[4], u1[4];
    float  bb[4];
    int    ncol[4];
#pragma unroll
    for (int j = 0; j < 4; ++j) {
        const int n = n0 + wn * 64 + j * 16 + l16;
        ncol[j] = n;
        const float4* up = (const float4*)(U + (size_t)n * R_DIM);
        u0[j] = up[0];
        u1[j] = up[1];
        bb[j] = bias[n];
    }

#pragma unroll
    for (int i = 0; i < 2; ++i) {
        const int mrowbase = m0 + wm * 32 + i * 16 + hihalf * 8;
#pragma unroll
        for (int e = 0; e < 8; ++e) {
            const int m = mrowbase + e;
            const float4* hp = (const float4*)(h + (size_t)m * R_DIM);
            float4 h0 = hp[0];
            float4 h1 = hp[1];
#pragma unroll
            for (int j = 0; j < 4; ++j) {
                float lora = h0.x * u0[j].x + h0.y * u0[j].y +
                             h0.z * u0[j].z + h0.w * u0[j].w +
                             h1.x * u1[j].x + h1.y * u1[j].y +
                             h1.z * u1[j].z + h1.w * u1[j].w;
                out[(size_t)m * N_DIM + ncol[j]] = acc[i][j][e] + bb[j] + lora;
            }
        }
    }
}

// ---------------------------------------------------------------------------
// one K-tile of compute: 24 ds_load_b128 + 16 WMMA (two K-slices of 32)
// Abase/Bbase already include the per-lane row and K half-offsets.
// ---------------------------------------------------------------------------
__device__ __forceinline__ void compute_tile(
    const unsigned short* Abase, const unsigned short* Bbase, v8f acc[2][4])
{
#pragma unroll
    for (int ks = 0; ks < 2; ++ks) {
        FragU a[2];
#pragma unroll
        for (int i = 0; i < 2; ++i) {
            const unsigned short* p = Abase + i * 16 * LDT2 + ks * 32;
            a[i].q[0] = *(const uint4v*)(p);
            a[i].q[1] = *(const uint4v*)(p + 16);
        }
        FragU b[4];
#pragma unroll
        for (int j = 0; j < 4; ++j) {
            const unsigned short* p = Bbase + j * 16 * LDT2 + ks * 32;
            b[j].q[0] = *(const uint4v*)(p);
            b[j].q[1] = *(const uint4v*)(p + 16);
        }
#pragma unroll
        for (int i = 0; i < 2; ++i)
#pragma unroll
            for (int j = 0; j < 4; ++j)
                acc[i][j] = __builtin_amdgcn_wmma_f32_16x16x32_bf16(
                    false, a[i].bf, false, b[j].bf,
                    (short)0, acc[i][j], false, false);
    }
}

// ---------------------------------------------------------------------------
// Kernel 2 (FAST): GEMM from pre-converted bf16 x / W.
// 128x128 block tile, BK2=64, double-buffered LDS filled by ASYNC
// global->LDS copies (no staging VGPRs), 256 threads (8 waves).
// ---------------------------------------------------------------------------
__global__ void __launch_bounds__(256)
gemm_lora_bf16_kernel(const unsigned short* __restrict__ xb,
                      const unsigned short* __restrict__ wb,
                      const float* __restrict__ bias, const float* __restrict__ U,
                      const float* __restrict__ h, float* __restrict__ out)
{
    __shared__ __align__(16) unsigned short sA[2][BM * LDT2];   // 2 x 18 KB
    __shared__ __align__(16) unsigned short sB[2][BN * LDT2];   // 2 x 18 KB

    const int tid    = threadIdx.x;
    const int lane   = tid & 31;
    const int wave   = tid >> 5;
    const int wm     = wave & 3;          // M strip: rows wm*32 .. +31
    const int wn     = wave >> 2;         // N strip: cols wn*64 .. +63
    const int hihalf = lane >> 4;
    const int l16    = lane & 15;

    const int m0 = blockIdx.y * BM;
    const int n0 = blockIdx.x * BN;

    // staging: thread t owns 32 contiguous bf16 (64 B) of one row
    const int srow = tid >> 1;            // 0..127
    const int scol = (tid & 1) * 32;      // 0 or 32 (bf16 units)
    const int soff = srow * LDT2 + scol;  // LDS element offset of this thread's 64 B

    const unsigned short* gA = xb + (size_t)(m0 + srow) * K_DIM + scol;
    const unsigned short* gB = wb + (size_t)(n0 + srow) * K_DIM + scol;

    v8f acc[2][4];
#pragma unroll
    for (int i = 0; i < 2; ++i)
#pragma unroll
        for (int j = 0; j < 4; ++j) acc[i][j] = (v8f){0,0,0,0,0,0,0,0};

    const int aKoff = hihalf * 8;         // A: K 0..7 / 8..15 (then +16)
    const int bKoff = hihalf * 8;         // B: K 0..7 / 8..15 (then +16)
    const unsigned short* A0 = &sA[0][(wm * 32 + l16) * LDT2 + aKoff];
    const unsigned short* A1 = &sA[1][(wm * 32 + l16) * LDT2 + aKoff];
    const unsigned short* B0 = &sB[0][(wn * 64 + l16) * LDT2 + bKoff];
    const unsigned short* B1 = &sB[1][(wn * 64 + l16) * LDT2 + bKoff];

    const int nk = K_DIM / BK2;           // 64 K-tiles (even)

    // ---- prologue: async-stage tile 0 into buffer 0 ----
    async_stage64B(gA, &sA[0][soff]);
    async_stage64B(gB, &sB[0][soff]);
    wait_async0();
    __syncthreads();

    for (int kt = 0; kt < nk; kt += 2) {
        // even tile: compute buf0, async-stage tile kt+1 -> buf1
        async_stage64B(gA + (size_t)(kt + 1) * BK2, &sA[1][soff]);
        async_stage64B(gB + (size_t)(kt + 1) * BK2, &sB[1][soff]);
        compute_tile(A0, B0, acc);
        wait_async0();
        __syncthreads();

        // odd tile: compute buf1, async-stage tile kt+2 -> buf0
        const bool more = (kt + 2) < nk;
        if (more) {
            async_stage64B(gA + (size_t)(kt + 2) * BK2, &sA[0][soff]);
            async_stage64B(gB + (size_t)(kt + 2) * BK2, &sB[0][soff]);
        }
        compute_tile(A1, B1, acc);
        if (more) {
            wait_async0();
            __syncthreads();
        }
    }

    gemm_epilogue(acc, bias, U, h, out, m0, n0, wm, wn, l16, hihalf);
}

// ---------------------------------------------------------------------------
// Kernel 2 (FALLBACK, ws too small): fp32 sources, inline conversion,
// single-buffered LDS, BK=32. Same math.
// ---------------------------------------------------------------------------
__global__ void __launch_bounds__(256)
gemm_lora_f32_kernel(const float* __restrict__ x, const float* __restrict__ W,
                     const float* __restrict__ bias, const float* __restrict__ U,
                     const float* __restrict__ h, float* __restrict__ out)
{
    __shared__ __align__(16) unsigned short sA[BM * LDT];
    __shared__ __align__(16) unsigned short sB[BN * LDT];

    const int tid    = threadIdx.x;
    const int lane   = tid & 31;
    const int wave   = tid >> 5;
    const int wm     = wave & 3;
    const int wn     = wave >> 2;
    const int hihalf = lane >> 4;
    const int l16    = lane & 15;

    const int m0 = blockIdx.y * BM;
    const int n0 = blockIdx.x * BN;

    const int srow = tid >> 1;
    const int scol = (tid & 1) * 16;

    v8f acc[2][4];
#pragma unroll
    for (int i = 0; i < 2; ++i)
#pragma unroll
        for (int j = 0; j < 4; ++j) acc[i][j] = (v8f){0,0,0,0,0,0,0,0};

    const int aRow0 = (wm * 32 + l16) * LDT;
    const int aKoff = hihalf * 8;
    const int bKoff = hihalf * 8;

    for (int k0 = 0; k0 < K_DIM; k0 += BK) {
        __syncthreads();
        {
            const float4* gp = (const float4*)(x + (size_t)(m0 + srow) * K_DIM + k0 + scol);
            float4 f0 = gp[0], f1 = gp[1], f2 = gp[2], f3 = gp[3];
            uint4v q0, q1;
            q0.x = pack_bf16(f0.x, f0.y);  q0.y = pack_bf16(f0.z, f0.w);
            q0.z = pack_bf16(f1.x, f1.y);  q0.w = pack_bf16(f1.z, f1.w);
            q1.x = pack_bf16(f2.x, f2.y);  q1.y = pack_bf16(f2.z, f2.w);
            q1.z = pack_bf16(f3.x, f3.y);  q1.w = pack_bf16(f3.z, f3.w);
            *(uint4v*)(&sA[srow * LDT + scol])     = q0;
            *(uint4v*)(&sA[srow * LDT + scol + 8]) = q1;
        }
        {
            const float4* gp = (const float4*)(W + (size_t)(n0 + srow) * K_DIM + k0 + scol);
            float4 f0 = gp[0], f1 = gp[1], f2 = gp[2], f3 = gp[3];
            uint4v q0, q1;
            q0.x = pack_bf16(f0.x, f0.y);  q0.y = pack_bf16(f0.z, f0.w);
            q0.z = pack_bf16(f1.x, f1.y);  q0.w = pack_bf16(f1.z, f1.w);
            q1.x = pack_bf16(f2.x, f2.y);  q1.y = pack_bf16(f2.z, f2.w);
            q1.z = pack_bf16(f3.x, f3.y);  q1.w = pack_bf16(f3.z, f3.w);
            *(uint4v*)(&sB[srow * LDT + scol])     = q0;
            *(uint4v*)(&sB[srow * LDT + scol + 8]) = q1;
        }
        __syncthreads();

        FragU a[2];
#pragma unroll
        for (int i = 0; i < 2; ++i) {
            const unsigned short* base = &sA[aRow0 + i * 16 * LDT + aKoff];
            a[i].q[0] = *(const uint4v*)(base);
            a[i].q[1] = *(const uint4v*)(base + 16);
        }
        FragU b[4];
#pragma unroll
        for (int j = 0; j < 4; ++j) {
            const unsigned short* base = &sB[(wn * 64 + j * 16 + l16) * LDT + bKoff];
            b[j].q[0] = *(const uint4v*)(base);
            b[j].q[1] = *(const uint4v*)(base + 16);
        }
#pragma unroll
        for (int i = 0; i < 2; ++i)
#pragma unroll
            for (int j = 0; j < 4; ++j)
                acc[i][j] = __builtin_amdgcn_wmma_f32_16x16x32_bf16(
                    false, a[i].bf, false, b[j].bf,
                    (short)0, acc[i][j], false, false);
    }

    gemm_epilogue(acc, bias, U, h, out, m0, n0, wm, wn, l16, hihalf);
}

// ---------------------------------------------------------------------------
extern "C" void kernel_launch(void* const* d_in, const int* in_sizes, int n_in,
                              void* d_out, int out_size, void* d_ws, size_t ws_size,
                              hipStream_t stream) {
    const float* x     = (const float*)d_in[0];   // [4,4096,4096]
    const float* codes = (const float*)d_in[1];   // [4,4096,8]
    const float* Wb    = (const float*)d_in[2];   // [4096,4096]
    const float* bb    = (const float*)d_in[3];   // [4096]
    const float* V     = (const float*)d_in[4];   // [4096,8]
    const float* U     = (const float*)d_in[5];   // [4096,8]
    float* out = (float*)d_out;                   // [4,4096,4096]

    const size_t h_bytes  = (size_t)M_DIM * R_DIM * sizeof(float);          // 512 KB
    const size_t xb_bytes = (size_t)M_DIM * K_DIM * sizeof(unsigned short); // 128 MB
    const size_t wb_bytes = (size_t)N_DIM * K_DIM * sizeof(unsigned short); //  32 MB

    float* h = (float*)d_ws;

    // 1) h = 2 * codes * (x @ V)
    hgen_kernel<<<M_DIM, 256, 0, stream>>>(x, codes, V, h);

    dim3 grid(N_DIM / BN, M_DIM / BM);

    if (ws_size >= h_bytes + xb_bytes + wb_bytes) {
        // fast path: one-time fp32->bf16 conversion, then async-staged WMMA GEMM
        unsigned short* xbf = (unsigned short*)((char*)d_ws + h_bytes);
        unsigned short* wbf = (unsigned short*)((char*)d_ws + h_bytes + xb_bytes);

        long nx8 = (long)M_DIM * K_DIM / 8;
        long nw8 = (long)N_DIM * K_DIM / 8;
        cvt_bf16_kernel<<<(unsigned)((nx8 + 255) / 256), 256, 0, stream>>>(x,  xbf, nx8);
        cvt_bf16_kernel<<<(unsigned)((nw8 + 255) / 256), 256, 0, stream>>>(Wb, wbf, nw8);

        gemm_lora_bf16_kernel<<<grid, 256, 0, stream>>>(xbf, wbf, bb, U, h, out);
    } else {
        // fallback: convert inside the GEMM
        gemm_lora_f32_kernel<<<grid, 256, 0, stream>>>(x, Wb, bb, U, h, out);
    }
}
